// PCFEncoder_58050777972736
// MI455X (gfx1250) — matrix operand order repaired and checked by reading, not compiled, not run on hardware
//
#include <hip/hip_runtime.h>
#include <hip/hip_bf16.h>

typedef __attribute__((ext_vector_type(16))) _Float16 v16h;
typedef __attribute__((ext_vector_type(8)))  float    v8f;

#define B_   8
#define N_   8192
#define M_   2000
#define K_   16
#define C1_  128
#define C2_  256
#define OUT_ 100
#define KP1  32    // layer-1 contraction dim padded 6 -> 32   (1 chunk)
#define KP2  160   // layer-2 contraction dim padded 131 -> 160 (5 chunks)
#define NCH2 (KP2 / 32)

// ---------------------------------------------------------------------------
// Pre-pack weights into WMMA *fragment-major* f16 panels so a B fragment is a
// single contiguous 32-byte (v16h) load per lane:
//   Wf[((tile*nchunk + chunk)*32 + lane)*16 + e] =
//       W_logical[k = chunk*32 + e + (lane>>4)*8 + (e>=8 ? 8 : 0)][col = tile*16 + (lane&15)]
// W1 logical rows: 0..5 = W1 (rel|ngh), rest zero.
// W2 logical rows: 0..127 = W2 rows 3..130 (f1 features first, so the A tile
//   row copy is vectorizable), 128..130 = W2 rows 0..2 (rel), rest zero.
// ---------------------------------------------------------------------------
__global__ void prep_weights(const float* __restrict__ W1,
                             const float* __restrict__ W2,
                             _Float16* __restrict__ W1f,   // [8*32*16]
                             _Float16* __restrict__ W2f) { // [16*5*32*16]
  const int total1 = (C1_ / 16) * 32 * 16;          // 4096
  const int total2 = (C2_ / 16) * NCH2 * 32 * 16;   // 40960
  for (int t = blockIdx.x * blockDim.x + threadIdx.x; t < total1 + total2;
       t += gridDim.x * blockDim.x) {
    if (t < total1) {
      int e = t & 15, L = (t >> 4) & 31, tile = t >> 9;
      int k = e + (L >> 4) * 8 + (e >= 8 ? 8 : 0);
      int col = tile * 16 + (L & 15);
      W1f[t] = (_Float16)(k < 6 ? W1[k * C1_ + col] : 0.0f);
    } else {
      int u = t - total1;
      int e = u & 15, L = (u >> 4) & 31;
      int chunk = (u >> 9) % NCH2, tile = u / (NCH2 * 512);
      int k = chunk * 32 + e + (L >> 4) * 8 + (e >= 8 ? 8 : 0);
      int col = tile * 16 + (L & 15);
      float v = 0.0f;
      if (k < 128)      v = W2[(3 + k) * C2_ + col];   // f1 features
      else if (k < 131) v = W2[(k - 128) * C2_ + col]; // rel coords
      W2f[u] = (_Float16)v;
    }
  }
}

// ---------------------------------------------------------------------------
// Farthest point sampling: one 1024-thread block per batch. Running
// min-distance array in LDS; argmax via LDS tree reduction. Serial over M
// (inherent), parallel over N and over the B batches.
// ---------------------------------------------------------------------------
__global__ void fps_kernel(const float* __restrict__ coords,  // [B][N][3]
                           int* __restrict__ fps_idx,         // [B][M]
                           float* __restrict__ sub) {         // [B][M][3]
  __shared__ float s_mind[N_];     // 32 KB
  __shared__ float r_val[1024];
  __shared__ int   r_idx[1024];
  __shared__ int   s_last;

  const int b = blockIdx.x, t = threadIdx.x, T = blockDim.x;
  const float* cb = coords + (size_t)b * N_ * 3;

  for (int i = t; i < N_; i += T) s_mind[i] = 1.0e30f;
  if (t == 0) s_last = 0;
  __syncthreads();

  for (int m = 0; m < M_; ++m) {
    const int last = s_last;
    const float lx = cb[last * 3 + 0];
    const float ly = cb[last * 3 + 1];
    const float lz = cb[last * 3 + 2];
    if (t == 0) {
      fps_idx[(size_t)b * M_ + m] = last;
      sub[((size_t)b * M_ + m) * 3 + 0] = lx;
      sub[((size_t)b * M_ + m) * 3 + 1] = ly;
      sub[((size_t)b * M_ + m) * 3 + 2] = lz;
    }
    if (m == M_ - 1) break;

    float bv = -1.0f; int bi = 0;
    for (int i = t; i < N_; i += T) {
      float dx = cb[i * 3 + 0] - lx;
      float dy = cb[i * 3 + 1] - ly;
      float dz = cb[i * 3 + 2] - lz;
      float d  = dx * dx + dy * dy + dz * dz;
      float mn = fminf(s_mind[i], d);
      s_mind[i] = mn;
      if (mn > bv || (mn == bv && i < bi)) { bv = mn; bi = i; }
    }
    r_val[t] = bv; r_idx[t] = bi;
    __syncthreads();
    for (int s = T >> 1; s > 0; s >>= 1) {
      if (t < s) {
        float ov = r_val[t + s]; int oi = r_idx[t + s];
        if (ov > r_val[t] || (ov == r_val[t] && oi < r_idx[t])) {
          r_val[t] = ov; r_idx[t] = oi;
        }
      }
      __syncthreads();
    }
    if (t == 0) s_last = r_idx[0];
    __syncthreads();
  }
}

// ---------------------------------------------------------------------------
// KNN: one wave per query. Per-lane register top-16 (unrolled compare-swap,
// stays in VGPRs), LDS merge + selection, global_prefetch ahead in the scan.
// Grid must be exactly (B*M)/8 blocks of 256 (16000/8 = 2000, exact).
// ---------------------------------------------------------------------------
__global__ void knn_kernel(const float* __restrict__ qpts,  // [B][M][3]
                           const float* __restrict__ spts,  // [B][nsrc][3]
                           int nsrc,
                           int* __restrict__ out_idx) {     // [B][M][K]
  __shared__ float s_d[8][512];
  __shared__ int   s_i[8][512];
  const int wave = threadIdx.x >> 5, lane = threadIdx.x & 31;
  const int qi = blockIdx.x * 8 + wave;
  const int b  = qi / M_;
  const float* qp = qpts + (size_t)qi * 3;
  const float* sb = spts + (size_t)b * nsrc * 3;
  const float qx = qp[0], qy = qp[1], qz = qp[2];

  float bd[K_]; int bi[K_];
#pragma unroll
  for (int t = 0; t < K_; ++t) { bd[t] = 3.0e38f; bi[t] = 0; }

  for (int j = lane; j < nsrc; j += 32) {
    if (j + 256 < nsrc) __builtin_prefetch(&sb[(j + 256) * 3], 0, 1);
    float dx = sb[j * 3 + 0] - qx;
    float dy = sb[j * 3 + 1] - qy;
    float dz = sb[j * 3 + 2] - qz;
    float d  = dx * dx + dy * dy + dz * dz;
    if (d < bd[K_ - 1]) {
      int jj = j;
#pragma unroll
      for (int t = 0; t < K_; ++t) {
        if (d < bd[t]) {
          float td = bd[t]; bd[t] = d;  d  = td;
          int   ti = bi[t]; bi[t] = jj; jj = ti;
        }
      }
    }
  }
#pragma unroll
  for (int t = 0; t < K_; ++t) {
    s_d[wave][lane * K_ + t] = bd[t];
    s_i[wave][lane * K_ + t] = bi[t];
  }
  __syncthreads();
  if (lane == 0) {
    for (int t = 0; t < K_; ++t) {
      float best = 3.9e38f; int bj = 0, bp = 0;
      for (int p = 0; p < 512; ++p) {
        float v = s_d[wave][p];
        if (v < best) { best = v; bj = s_i[wave][p]; bp = p; }
      }
      s_d[wave][bp] = 3.9e38f;
      out_idx[(size_t)qi * K_ + t] = bj;
    }
  }
}

// ---------------------------------------------------------------------------
// A tile is stored in LDS in *fragment-major* order:
//   sA[(chunk*32 + L)*16 + e] = A[row = L&15][k = chunk*32 + kmap(e, L>>4)]
// with kmap(e,h) = e + 8h + (e>=8 ? 8 : 0). For fixed (chunk,h) the k-indices
// form two contiguous 8-half runs, so the builder writes 16-byte chunks
// (ds_store_b128) and every reader lane loads 32 contiguous bytes
// (2x ds_load_b128) per fragment.
// ---------------------------------------------------------------------------

// Layer 1: per query (one wave): A = 16 neighbors x KP1 (rel|ngh|0...),
// 8 column tiles of v_wmma_f32_16x16x32_f16; bias+relu+max-over-K via
// in-lane max + shfl_xor(16). EXEC all-ones at every WMMA.
__global__ void layer1_kernel(const float* __restrict__ coords,  // [B][N][3]
                              const float* __restrict__ sub,     // [B][M][3]
                              const int* __restrict__ idx1,      // [B][M][K]
                              const _Float16* __restrict__ W1f,  // fragment-major
                              const float* __restrict__ b1,      // [C1]
                              _Float16* __restrict__ f1h) {      // [B][M][C1]
  __shared__ alignas(32) _Float16 sA[8][32 * 16];  // 8 KB
  const int wave = threadIdx.x >> 5, lane = threadIdx.x & 31;
  const int qi = blockIdx.x * 8 + wave;
  const int b  = qi / M_;
  const float* sp = sub + (size_t)qi * 3;
  const float sx = sp[0], sy = sp[1], sz = sp[2];

  if (lane < K_) {
    int n = idx1[(size_t)qi * K_ + lane];
    const float* np = coords + ((size_t)b * N_ + n) * 3;
    float nx = np[0], ny = np[1], nz = np[2];
    _Float16 fr[32];
#pragma unroll
    for (int c = 0; c < 32; ++c) fr[c] = (_Float16)0.0f;
    fr[0] = (_Float16)(nx - sx);
    fr[1] = (_Float16)(ny - sy);
    fr[2] = (_Float16)(nz - sz);
    fr[3] = (_Float16)nx;
    fr[4] = (_Float16)ny;
    fr[5] = (_Float16)nz;
#pragma unroll
    for (int h = 0; h < 2; ++h) {
      _Float16* dst = &sA[wave][(h * 16 + lane) * 16];
      __builtin_memcpy(dst,     &fr[8 * h],      16);
      __builtin_memcpy(dst + 8, &fr[8 * h + 16], 16);
    }
  }
  __syncthreads();

  const int r = lane & 15;
  v16h a;
  __builtin_memcpy(&a, &sA[wave][lane * 16], 32);   // 2x ds_load_b128

#pragma unroll 2
  for (int tile = 0; tile < C1_ / 16; ++tile) {
    const int col = tile * 16 + r;
    v16h bf;
    __builtin_memcpy(&bf, &W1f[(size_t)(tile * 32 + lane) * 16], 32);
    v8f c = {};
    c = __builtin_amdgcn_wmma_f32_16x16x32_f16(false, a, false, bf,
                                               (short)0, c, false, false);
    const float bias = b1[col];
    float mx = -3.0e38f;
#pragma unroll
    for (int v = 0; v < 8; ++v) {
      float x = c[v] + bias;
      x = x > 0.0f ? x : 0.0f;                 // relu
      mx = fmaxf(mx, x);
    }
    mx = fmaxf(mx, __shfl_xor(mx, 16, 32));    // rows 0..7 <-> 8..15
    if (lane < 16) f1h[(size_t)qi * C1_ + col] = (_Float16)mx;
  }
}

// Layer 2: per query A = 16 neighbors x KP2 (f1[128] | rel[3] | 0...).
// A fragments (5 chunks) hoisted out of the tile loop; inner loop is
// 5 B loads (v16h -> 2x global_load_b128) + 5 WMMAs + epilogue.
__global__ void layer2_kernel(const float* __restrict__ sub,     // [B][M][3]
                              const int* __restrict__ idx2,      // [B][M][K]
                              const _Float16* __restrict__ f1h,  // [B][M][C1]
                              const _Float16* __restrict__ W2f,  // fragment-major
                              const float* __restrict__ b2,      // [C2]
                              float* __restrict__ f2) {          // [B][M][C2]
  __shared__ alignas(32) _Float16 sA[8][NCH2 * 32 * 16];  // 40 KB
  const int wave = threadIdx.x >> 5, lane = threadIdx.x & 31;
  const int qi = blockIdx.x * 8 + wave;
  const int b  = qi / M_;
  const float* sp = sub + (size_t)qi * 3;

  if (lane < K_) {
    int n = idx2[(size_t)qi * K_ + lane];
    const float* np = sub + ((size_t)b * M_ + n) * 3;
    _Float16 fr[KP2];
    const _Float16* src = f1h + ((size_t)b * M_ + n) * C1_;
#pragma unroll
    for (int w = 0; w < C1_ / 8; ++w)          // 16 x 16-byte row copies
      __builtin_memcpy(&fr[w * 8], &src[w * 8], 16);
    fr[128] = (_Float16)(np[0] - sp[0]);
    fr[129] = (_Float16)(np[1] - sp[1]);
    fr[130] = (_Float16)(np[2] - sp[2]);
#pragma unroll
    for (int c = 131; c < KP2; ++c) fr[c] = (_Float16)0.0f;
#pragma unroll
    for (int chunk = 0; chunk < NCH2; ++chunk) {
#pragma unroll
      for (int h = 0; h < 2; ++h) {
        _Float16* dst = &sA[wave][(chunk * 32 + h * 16 + lane) * 16];
        __builtin_memcpy(dst,     &fr[chunk * 32 + 8 * h],      16);
        __builtin_memcpy(dst + 8, &fr[chunk * 32 + 8 * h + 16], 16);
      }
    }
  }
  __syncthreads();

  const int r = lane & 15;
  v16h afrag[NCH2];
#pragma unroll
  for (int chunk = 0; chunk < NCH2; ++chunk)
    __builtin_memcpy(&afrag[chunk], &sA[wave][(chunk * 32 + lane) * 16], 32);

#pragma unroll 2
  for (int tile = 0; tile < C2_ / 16; ++tile) {
    const int col = tile * 16 + r;
    v8f c = {};
#pragma unroll
    for (int chunk = 0; chunk < NCH2; ++chunk) {
      v16h bf;
      __builtin_memcpy(&bf,
                       &W2f[(size_t)((tile * NCH2 + chunk) * 32 + lane) * 16],
                       32);
      c = __builtin_amdgcn_wmma_f32_16x16x32_f16(false, afrag[chunk], false,
                                                 bf, (short)0, c, false, false);
    }
    const float bias = b2[col];
    float mx = -3.0e38f;
#pragma unroll
    for (int v = 0; v < 8; ++v) {
      float x = c[v] + bias;
      x = x > 0.0f ? x : 0.0f;
      mx = fmaxf(mx, x);
    }
    mx = fmaxf(mx, __shfl_xor(mx, 16, 32));
    if (lane < 16) f2[(size_t)qi * C2_ + col] = mx;
  }
}

// ---------------------------------------------------------------------------
// Fused adaptive max+avg pool over uniform windows of M/OUT = 20.
// out[b][c][o] = max_j f2[b][o*20+j][c] + mean_j f2[b][o*20+j][c]
// ---------------------------------------------------------------------------
__global__ void pool_kernel(const float* __restrict__ f2,
                            float* __restrict__ out) {
  const int tid = blockIdx.x * blockDim.x + threadIdx.x;
  if (tid >= B_ * C2_ * OUT_) return;
  const int o = tid % OUT_;
  const int c = (tid / OUT_) % C2_;
  const int b = tid / (OUT_ * C2_);
  const int W = M_ / OUT_;  // 20
  float mx = -3.0e38f, sm = 0.0f;
  for (int j = 0; j < W; ++j) {
    float v = f2[((size_t)b * M_ + o * W + j) * C2_ + c];
    mx = fmaxf(mx, v);
    sm += v;
  }
  out[tid] = mx + sm / (float)W;
}

// ---------------------------------------------------------------------------
extern "C" void kernel_launch(void* const* d_in, const int* in_sizes, int n_in,
                              void* d_out, int out_size, void* d_ws,
                              size_t ws_size, hipStream_t stream) {
  (void)in_sizes; (void)n_in; (void)out_size; (void)ws_size;
  const float* x  = (const float*)d_in[0];
  const float* W1 = (const float*)d_in[1];
  const float* b1 = (const float*)d_in[2];
  const float* W2 = (const float*)d_in[3];
  const float* b2 = (const float*)d_in[4];
  float* out = (float*)d_out;

  char* ws = (char*)d_ws;
  size_t off = 0;
  auto wsalloc = [&](size_t bytes) -> char* {
    char* p = ws + off;
    off = (off + bytes + 255) & ~(size_t)255;
    return p;
  };
  _Float16* W1f    = (_Float16*)wsalloc((size_t)(C1_ / 16) * 32 * 16 * 2);
  _Float16* W2f    = (_Float16*)wsalloc((size_t)(C2_ / 16) * NCH2 * 32 * 16 * 2);
  int*      fpsidx = (int*)     wsalloc((size_t)B_ * M_ * 4);
  float*    sub    = (float*)   wsalloc((size_t)B_ * M_ * 3 * 4);
  int*      idx1   = (int*)     wsalloc((size_t)B_ * M_ * K_ * 4);
  int*      idx2   = (int*)     wsalloc((size_t)B_ * M_ * K_ * 4);
  _Float16* f1h    = (_Float16*)wsalloc((size_t)B_ * M_ * C1_ * 2);
  float*    f2     = (float*)   wsalloc((size_t)B_ * M_ * C2_ * 4);

  const int qblocks = (B_ * M_) / 8;  // 2000, exact

  prep_weights<<<64, 256, 0, stream>>>(W1, W2, W1f, W2f);
  fps_kernel<<<B_, 1024, 0, stream>>>(x, fpsidx, sub);
  knn_kernel<<<qblocks, 256, 0, stream>>>(sub, x, N_, idx1);
  layer1_kernel<<<qblocks, 256, 0, stream>>>(x, sub, idx1, W1f, b1, f1h);
  knn_kernel<<<qblocks, 256, 0, stream>>>(sub, sub, M_, idx2);
  layer2_kernel<<<qblocks, 256, 0, stream>>>(sub, idx2, f1h, W2f, b2, f2);
  pool_kernel<<<(B_ * C2_ * OUT_ + 255) / 256, 256, 0, stream>>>(f2, out);
}
